// Tacotron_3195455668664
// MI455X (gfx1250) — matrix operand order repaired and checked by reading, not compiled, not run on hardware
//
#include <hip/hip_runtime.h>
#include <cstdint>
#include <cstddef>

// ---------------- problem constants ----------------
#define B_    32
#define TENC  200
#define E_    512
#define D_    1024
#define P_    256
#define M_    80
#define TDEC  600
#define A_    128
#define LF_   32
#define KW_   31
#define KA_   1792   // P_+E_+D_  (att LSTM concat K)
#define KG_   2560   // D_+E_+D_  (gen LSTM concat K)
#define G4_   4096   // 4*D_
#define NWG_  96
#define BLK_  256
#define NWAVE_ (NWG_*(BLK_/32))

typedef __attribute__((ext_vector_type(16))) __bf16  v16bf;
typedef __attribute__((ext_vector_type(8)))  float   v8f;
typedef __attribute__((ext_vector_type(4)))  unsigned v4u;

union FragBF { v16bf bf; v4u u[2]; };

struct Wsp {
  unsigned* ctrl;
  __bf16 *Watt, *Wgen, *Wloc, *W1p, *W2p, *tgtb, *x1b, *preb;
  float*  proc;
  __bf16 *xatt, *xgen, *locf;
  float  *g1, *g2, *locp, *hatt, *catt, *hgen, *cgen, *ctx, *wprev, *wcum;
};
struct Ins {
  const float* enc; const int* lens; const float* target;
  const float *pw1,*pb1,*pw2,*pb2;
  const float *attWih,*attWhh,*attb,*genWih,*genWhh,*genb;
  const float *Wq,*Wm,*lock,*Wlocf,*vatt,*batt,*frameW,*frameb,*stopW,*stopb;
};
struct Outs { float *spec, *stop, *align; };

// ---------------- device helpers ----------------
__device__ __forceinline__ float sigf(float x) { return 1.0f / (1.0f + expf(-x)); }

// Single 16x16 output tile.  A[16 x K] row-major bf16 (lda = K); B = 16 weight
// rows, each K contiguous bf16.  Both fragments are two contiguous 16B loads
// per lane, matching the CDNA5 16-bit A / B WMMA VGPR layouts.
__device__ __forceinline__ v8f tileK(const __bf16* __restrict__ arow,
                                     const __bf16* __restrict__ brow,
                                     int K, int half) {
  v8f acc = {0.f,0.f,0.f,0.f,0.f,0.f,0.f,0.f};
  for (int kb = 0; kb < K; kb += 32) {
    __builtin_prefetch(brow + kb + 512, 0, 1);
    FragBF a, b;
    a.u[0] = *(const v4u*)(arow + kb + half * 8);
    a.u[1] = *(const v4u*)(arow + kb + 16 + half * 8);
    b.u[0] = *(const v4u*)(brow + kb + half * 8);
    b.u[1] = *(const v4u*)(brow + kb + 16 + half * 8);
    acc = __builtin_amdgcn_wmma_f32_16x16x32_bf16(
        false, a.bf, false, b.bf, (short)0, acc, false, false);
  }
  return acc;
}

// Two M-tiles (rows mb0.. and mb1..) sharing one B fragment: halves weight
// traffic (weights dominate per-step bytes) and doubles WMMA density per
// loop body for better latency hiding.
__device__ __forceinline__ void tileK2(const __bf16* __restrict__ a0row,
                                       const __bf16* __restrict__ a1row,
                                       const __bf16* __restrict__ brow,
                                       int K, int half, v8f& acc0, v8f& acc1) {
  for (int kb = 0; kb < K; kb += 32) {
    __builtin_prefetch(brow + kb + 512, 0, 1);
    FragBF a0, a1, b;
    b.u[0]  = *(const v4u*)(brow + kb + half * 8);
    b.u[1]  = *(const v4u*)(brow + kb + 16 + half * 8);
    a0.u[0] = *(const v4u*)(a0row + kb + half * 8);
    a0.u[1] = *(const v4u*)(a0row + kb + 16 + half * 8);
    a1.u[0] = *(const v4u*)(a1row + kb + half * 8);
    a1.u[1] = *(const v4u*)(a1row + kb + 16 + half * 8);
    acc0 = __builtin_amdgcn_wmma_f32_16x16x32_bf16(
        false, a0.bf, false, b.bf, (short)0, acc0, false, false);
    acc1 = __builtin_amdgcn_wmma_f32_16x16x32_bf16(
        false, a1.bf, false, b.bf, (short)0, acc1, false, false);
  }
}

// D layout: lane half h, lane&15 = column; VGPR v -> row h*8+v
__device__ __forceinline__ void storeF(float* __restrict__ C, int ldc,
                                       int mb, int nb, int lane, v8f acc) {
  const int half = lane >> 4, l16 = lane & 15;
  float* p = C + (size_t)(mb + half * 8) * ldc + nb + l16;
#pragma unroll
  for (int v = 0; v < 8; ++v) p[(size_t)v * ldc] = acc[v];
}

// grid-wide barrier (persistent kernel; all NWG_ blocks co-resident)
__device__ __forceinline__ void gbar(unsigned* cnt, unsigned& barn) {
  barn++;
  __threadfence();
  __syncthreads();
  if (threadIdx.x == 0) {
    __hip_atomic_fetch_add(cnt, 1u, __ATOMIC_RELEASE, __HIP_MEMORY_SCOPE_AGENT);
    const unsigned tgt = barn * (unsigned)NWG_;
    while (__hip_atomic_load(cnt, __ATOMIC_ACQUIRE, __HIP_MEMORY_SCOPE_AGENT) < tgt)
      __builtin_amdgcn_s_sleep(2);
  }
  __syncthreads();
  __threadfence();
}

__device__ __forceinline__ void frame_stop(const Wsp& w, const Ins& in,
                                           const Outs& out, int t,
                                           int gtid, int nthreads) {
  for (int i = gtid; i < B_ * M_ + B_; i += nthreads) {
    if (i < B_ * M_) {
      const int b = i / M_, m = i % M_;
      const float* fw = in.frameW + (size_t)m * (D_ + E_);
      const float* hg = w.hgen + (size_t)b * D_;
      const float* cx = w.ctx + (size_t)b * E_;
      float acc = in.frameb[m];
      for (int k = 0; k < D_; ++k) acc += fw[k] * hg[k];
      for (int k = 0; k < E_; ++k) acc += fw[D_ + k] * cx[k];
      out.spec[((size_t)b * TDEC + t) * M_ + m] = acc;
    } else {
      const int b = i - B_ * M_;
      const float* hg = w.hgen + (size_t)b * D_;
      const float* cx = w.ctx + (size_t)b * E_;
      float acc = in.stopb[0];
      for (int k = 0; k < D_; ++k) acc += in.stopW[k] * hg[k];
      for (int k = 0; k < E_; ++k) acc += in.stopW[D_ + k] * cx[k];
      out.stop[(size_t)b * TDEC + t] = acc;
    }
  }
}

// ---------------- one-time precompute kernels ----------------
__global__ void k_init(Wsp w) {
  const long gtid = (long)blockIdx.x * blockDim.x + threadIdx.x;
  const long nt = (long)gridDim.x * blockDim.x;
  if (gtid == 0) *w.ctrl = 0u;
  for (long i = gtid; i < B_ * KA_; i += nt) w.xatt[i] = (__bf16)0.f;
  for (long i = gtid; i < B_ * KG_; i += nt) w.xgen[i] = (__bf16)0.f;
  for (long i = gtid; i < B_ * D_; i += nt) { w.hatt[i]=0.f; w.catt[i]=0.f; w.hgen[i]=0.f; w.cgen[i]=0.f; }
  for (long i = gtid; i < B_ * E_; i += nt) w.ctx[i] = 0.f;
  for (long i = gtid; i < B_ * TENC; i += nt) { w.wprev[i]=0.f; w.wcum[i]=0.f; }
}

__global__ void k_convert(Wsp w, Ins in) {
  const long N1 = (long)G4_ * KA_, N2 = (long)G4_ * KG_;
  const long N3 = A_ * LF_, N4 = P_ * 96, N5 = P_ * P_;
  const long total = N1 + N2 + N3 + N4 + N5;
  const long nt = (long)gridDim.x * blockDim.x;
  for (long i = (long)blockIdx.x * blockDim.x + threadIdx.x; i < total; i += nt) {
    long j = i;
    if (j < N1) {                       // Watt = [att_Wih | att_Whh], rows=4096, K=1792
      const int n = (int)(j / KA_), k = (int)(j % KA_);
      const float v = (k < 768) ? in.attWih[(long)n * 768 + k]
                                : in.attWhh[(long)n * 1024 + (k - 768)];
      w.Watt[j] = (__bf16)v; continue;
    } j -= N1;
    if (j < N2) {                       // Wgen = [gen_Wih | gen_Whh], K=2560
      const int n = (int)(j / KG_), k = (int)(j % KG_);
      const float v = (k < 1536) ? in.genWih[(long)n * 1536 + k]
                                 : in.genWhh[(long)n * 1024 + (k - 1536)];
      w.Wgen[j] = (__bf16)v; continue;
    } j -= N2;
    if (j < N3) { w.Wloc[j] = (__bf16)in.Wlocf[j]; continue; } j -= N3;
    if (j < N4) {                       // prenet W1 padded 80 -> 96 in K
      const int n = (int)(j / 96), k = (int)(j % 96);
      w.W1p[j] = (__bf16)((k < 80) ? in.pw1[(long)n * 80 + k] : 0.f); continue;
    } j -= N4;
    w.W2p[j] = (__bf16)in.pw2[j];
  }
}

__global__ void k_build_tgt(Wsp w, Ins in) {   // tgt_bf[19200][96] (zero row 0, shift-by-1)
  const long total = (long)B_ * TDEC * 96;
  const long nt = (long)gridDim.x * blockDim.x;
  for (long i = (long)blockIdx.x * blockDim.x + threadIdx.x; i < total; i += nt) {
    const int r = (int)(i / 96), k = (int)(i % 96);
    const int b = r / TDEC, j = r % TDEC;
    float v = 0.f;
    if (k < M_ && j > 0) v = in.target[((long)b * M_ + k) * TDEC + (j - 1)];
    w.tgtb[i] = (__bf16)v;
  }
}

__global__ void k_procmem(Wsp w, Ins in) {     // proc_mem[6400][128] = enc @ Wm.T  (one-time f32)
  const long total = (long)B_ * TENC * A_;
  const long nt = (long)gridDim.x * blockDim.x;
  for (long i = (long)blockIdx.x * blockDim.x + threadIdx.x; i < total; i += nt) {
    const int a = (int)(i & (A_ - 1)); const int r = (int)(i >> 7);
    const float* er = in.enc + (size_t)r * E_;
    const float* wm = in.Wm + (size_t)a * E_;
    float acc = 0.f;
    for (int e = 0; e < E_; ++e) acc += er[e] * wm[e];
    w.proc[i] = acc;
  }
}

__global__ void k_prenet1(Wsp w, Ins in) {     // [19200x96] @ W1p -> relu -> x1b bf16
  const int lane = threadIdx.x & 31, half = lane >> 4, l16 = lane & 15;
  const int gw = (int)((blockIdx.x * blockDim.x + threadIdx.x) >> 5);
  const int tw = (int)((gridDim.x * blockDim.x) >> 5);
  for (int job = gw; job < (B_ * TDEC / 16) * (P_ / 16); job += tw) {
    const int mb = (job >> 4) * 16, nb = (job & 15) * 16;
    v8f acc = tileK(w.tgtb + (size_t)(mb + l16) * 96,
                    w.W1p + (size_t)(nb + l16) * 96, 96, half);
    const int col = nb + l16;
    const float bias = in.pb1[col];
#pragma unroll
    for (int v = 0; v < 8; ++v) {
      const int r = mb + half * 8 + v;
      w.x1b[(size_t)r * P_ + col] = (__bf16)fmaxf(acc[v] + bias, 0.f);
    }
  }
}

__global__ void k_prenet2(Wsp w, Ins in) {     // x1b @ W2p -> relu -> preb[t][b][256] bf16
  const int lane = threadIdx.x & 31, half = lane >> 4, l16 = lane & 15;
  const int gw = (int)((blockIdx.x * blockDim.x + threadIdx.x) >> 5);
  const int tw = (int)((gridDim.x * blockDim.x) >> 5);
  for (int job = gw; job < (B_ * TDEC / 16) * (P_ / 16); job += tw) {
    const int mb = (job >> 4) * 16, nb = (job & 15) * 16;
    v8f acc = tileK(w.x1b + (size_t)(mb + l16) * P_,
                    w.W2p + (size_t)(nb + l16) * P_, P_, half);
    const int col = nb + l16;
    const float bias = in.pb2[col];
#pragma unroll
    for (int v = 0; v < 8; ++v) {
      const int r = mb + half * 8 + v;
      const int b = r / TDEC, j = r % TDEC;
      w.preb[((size_t)j * B_ + b) * P_ + col] = (__bf16)fmaxf(acc[v] + bias, 0.f);
    }
  }
}

// ---------------- persistent decoder (whole 600-step scan) ----------------
__global__ __launch_bounds__(BLK_) void k_decoder(Wsp w, Ins in, Outs out) {
  __shared__ float q_s[BLK_ / 32][A_];
  __shared__ float w_s[BLK_ / 32][TENC];
  const int tid = threadIdx.x, lane = tid & 31, wv = tid >> 5;
  const int half = lane >> 4, l16 = lane & 15;
  const int gtid = blockIdx.x * BLK_ + tid;
  const int gw = gtid >> 5;
  const int nthreads = NWG_ * BLK_;
  unsigned barn = 0;

  for (int t = 0; t < TDEC; ++t) {
    // ---- P0: prenet copy + location conv + frame/stop of step t-1 ----
    for (int i = gtid; i < B_ * P_; i += nthreads) {
      const int b = i >> 8, p = i & 255;
      w.xatt[(size_t)b * KA_ + p] = w.preb[((size_t)t * B_ + b) * P_ + p];
    }
    for (int i = gtid; i < B_ * TENC * LF_; i += nthreads) {
      const int lf = i & 31; const int r = i >> 5;
      const int b = r / TENC, tp = r % TENC;
      float acc = 0.f;
      for (int k = 0; k < KW_; ++k) {
        const int tk = tp + k - 15;
        if (tk >= 0 && tk < TENC) {
          acc += w.wprev[b * TENC + tk] * in.lock[(lf * 2 + 0) * KW_ + k];
          acc += w.wcum [b * TENC + tk] * in.lock[(lf * 2 + 1) * KW_ + k];
        }
      }
      w.locf[(size_t)r * LF_ + lf] = (__bf16)acc;
    }
    if (t > 0) frame_stop(w, in, out, t - 1, gtid, nthreads);
    gbar(w.ctrl, barn);

    // ---- P1: att-LSTM gates GEMM (256 dual-M jobs, K=1792)
    //        + loc-proj GEMM (1600 dual-M jobs, K=32) ----
    {
      const int J1 = G4_ / 16;                       // 256
      const int J2 = (B_ * TENC / 32) * (A_ / 16);   // 200*8 = 1600
      for (int job = gw; job < J1 + J2; job += NWAVE_) {
        if (job < J1) {
          const int nb = job * 16;
          v8f acc0 = {0.f,0.f,0.f,0.f,0.f,0.f,0.f,0.f};
          v8f acc1 = {0.f,0.f,0.f,0.f,0.f,0.f,0.f,0.f};
          tileK2(w.xatt + (size_t)l16 * KA_,
                 w.xatt + (size_t)(16 + l16) * KA_,
                 w.Watt + (size_t)(nb + l16) * KA_, KA_, half, acc0, acc1);
          storeF(w.g1, G4_, 0, nb, lane, acc0);
          storeF(w.g1, G4_, 16, nb, lane, acc1);
        } else {
          const int j = job - J1;
          const int mb = (j >> 3) * 32, nb = (j & 7) * 16;
          v8f acc0 = {0.f,0.f,0.f,0.f,0.f,0.f,0.f,0.f};
          v8f acc1 = {0.f,0.f,0.f,0.f,0.f,0.f,0.f,0.f};
          tileK2(w.locf + (size_t)(mb + l16) * LF_,
                 w.locf + (size_t)(mb + 16 + l16) * LF_,
                 w.Wloc + (size_t)(nb + l16) * LF_, LF_, half, acc0, acc1);
          storeF(w.locp, A_, mb, nb, lane, acc0);
          storeF(w.locp, A_, mb + 16, nb, lane, acc1);
        }
      }
    }
    gbar(w.ctrl, barn);

    // ---- P2: att-LSTM pointwise ----
    for (int i = gtid; i < B_ * D_; i += nthreads) {
      const int b = i >> 10, n = i & 1023;
      const float* g = w.g1 + (size_t)b * G4_;
      const float gi = g[n]        + in.attb[n];
      const float gf = g[n + 1024] + in.attb[n + 1024];
      const float gg = g[n + 2048] + in.attb[n + 2048];
      const float go = g[n + 3072] + in.attb[n + 3072];
      const float c = sigf(gf) * w.catt[i] + sigf(gi) * tanhf(gg);
      const float h = sigf(go) * tanhf(c);
      w.catt[i] = c; w.hatt[i] = h;
      const __bf16 hb = (__bf16)h;
      w.xatt[(size_t)b * KA_ + (P_ + E_) + n] = hb;   // next step's att gates
      w.xgen[(size_t)b * KG_ + n] = hb;               // this step's gen gates
    }
    gbar(w.ctrl, barn);

    // ---- P4: attention (one wave per batch row): query, e, softmax, context ----
    {
      const bool act = gw < B_;
      const int b = gw;
      if (act) {
        const float* hr = w.hatt + (size_t)b * D_;
        for (int a = lane; a < A_; a += 32) {
          const float* wq = in.Wq + (size_t)a * D_;
          float q = 0.f;
          for (int k = 0; k < D_; ++k) q += hr[k] * wq[k];
          q_s[wv][a] = q;
        }
      }
      __syncthreads();
      float pmax = -3.0e38f, psum = 0.f;
      const int len = act ? in.lens[b] : 0;
      if (act) {
        for (int tp = lane; tp < TENC; tp += 32) {
          float ev = -1.0e9f;
          if (tp < len) {
            const float* pm = w.proc + ((size_t)b * TENC + tp) * A_;
            const float* lp = w.locp + ((size_t)b * TENC + tp) * A_;
            float acc = 0.f;
            for (int a = 0; a < A_; ++a)
              acc += tanhf(q_s[wv][a] + pm[a] + lp[a] + in.batt[a]) * in.vatt[a];
            ev = acc;
          }
          w_s[wv][tp] = ev;
          pmax = fmaxf(pmax, ev);
        }
      }
      for (int m = 16; m > 0; m >>= 1) pmax = fmaxf(pmax, __shfl_xor(pmax, m, 32));
      if (act) {
        for (int tp = lane; tp < TENC; tp += 32) {
          const float p = expf(w_s[wv][tp] - pmax);
          w_s[wv][tp] = p; psum += p;
        }
      }
      for (int m = 16; m > 0; m >>= 1) psum += __shfl_xor(psum, m, 32);
      if (act) {
        const float inv = 1.f / psum;
        for (int tp = lane; tp < TENC; tp += 32) {
          const float wt = w_s[wv][tp] * inv;
          w_s[wv][tp] = wt;
          w.wprev[b * TENC + tp] = wt;
          w.wcum[b * TENC + tp] += wt;
          out.align[((size_t)b * TDEC + t) * TENC + tp] = wt;
        }
      }
      __syncthreads();
      if (act) {
        for (int e = lane; e < E_; e += 32) {
          const float* er = in.enc + (size_t)b * TENC * E_ + e;
          float acc = 0.f;
          for (int tp = 0; tp < TENC; ++tp) acc += w_s[wv][tp] * er[(size_t)tp * E_];
          w.ctx[(size_t)b * E_ + e] = acc;
          const __bf16 cb = (__bf16)acc;
          w.xatt[(size_t)b * KA_ + P_ + e] = cb;      // next step att gates
          w.xgen[(size_t)b * KG_ + D_ + e] = cb;      // this step gen gates
        }
      }
    }
    gbar(w.ctrl, barn);

    // ---- P5: gen-LSTM gates GEMM (256 dual-M jobs, K=2560) ----
    for (int job = gw; job < G4_ / 16; job += NWAVE_) {
      const int nb = job * 16;
      v8f acc0 = {0.f,0.f,0.f,0.f,0.f,0.f,0.f,0.f};
      v8f acc1 = {0.f,0.f,0.f,0.f,0.f,0.f,0.f,0.f};
      tileK2(w.xgen + (size_t)l16 * KG_,
             w.xgen + (size_t)(16 + l16) * KG_,
             w.Wgen + (size_t)(nb + l16) * KG_, KG_, half, acc0, acc1);
      storeF(w.g2, G4_, 0, nb, lane, acc0);
      storeF(w.g2, G4_, 16, nb, lane, acc1);
    }
    gbar(w.ctrl, barn);

    // ---- P6: gen-LSTM pointwise ----
    for (int i = gtid; i < B_ * D_; i += nthreads) {
      const int b = i >> 10, n = i & 1023;
      const float* g = w.g2 + (size_t)b * G4_;
      const float gi = g[n]        + in.genb[n];
      const float gf = g[n + 1024] + in.genb[n + 1024];
      const float gg = g[n + 2048] + in.genb[n + 2048];
      const float go = g[n + 3072] + in.genb[n + 3072];
      const float c = sigf(gf) * w.cgen[i] + sigf(gi) * tanhf(gg);
      const float h = sigf(go) * tanhf(c);
      w.cgen[i] = c; w.hgen[i] = h;
      w.xgen[(size_t)b * KG_ + (D_ + E_) + n] = (__bf16)h;  // next step gen gates
    }
    gbar(w.ctrl, barn);
  }
  frame_stop(w, in, out, TDEC - 1, gtid, nthreads);
}

// ---------------- host launch ----------------
extern "C" void kernel_launch(void* const* d_in, const int* in_sizes, int n_in,
                              void* d_out, int out_size, void* d_ws, size_t ws_size,
                              hipStream_t stream) {
  (void)in_sizes; (void)n_in; (void)out_size; (void)ws_size;
  Ins in;
  in.enc    = (const float*)d_in[0];  in.lens  = (const int*)d_in[1];
  in.target = (const float*)d_in[2];  // d_in[3] = teacher_forcing_ratio (==1, unused)
  in.pw1 = (const float*)d_in[4];  in.pb1 = (const float*)d_in[5];
  in.pw2 = (const float*)d_in[6];  in.pb2 = (const float*)d_in[7];
  in.attWih = (const float*)d_in[8];  in.attWhh = (const float*)d_in[9];
  in.attb   = (const float*)d_in[10];
  in.genWih = (const float*)d_in[11]; in.genWhh = (const float*)d_in[12];
  in.genb   = (const float*)d_in[13];
  in.Wq = (const float*)d_in[14]; in.Wm = (const float*)d_in[15];
  in.lock = (const float*)d_in[16]; in.Wlocf = (const float*)d_in[17];
  in.vatt = (const float*)d_in[18]; in.batt = (const float*)d_in[19];
  in.frameW = (const float*)d_in[20]; in.frameb = (const float*)d_in[21];
  in.stopW  = (const float*)d_in[22]; in.stopb  = (const float*)d_in[23];

  Outs out;
  out.spec  = (float*)d_out;
  out.stop  = out.spec + (size_t)B_ * TDEC * M_;
  out.align = out.stop + (size_t)B_ * TDEC;

  // workspace carve-out (~68 MB)
  char* p = (char*)d_ws;
  auto take = [&](size_t bytes) { char* r = p; p += (bytes + 255) & ~(size_t)255; return r; };
  Wsp w;
  w.ctrl = (unsigned*)take(256);
  w.Watt = (__bf16*)take((size_t)G4_ * KA_ * 2);
  w.Wgen = (__bf16*)take((size_t)G4_ * KG_ * 2);
  w.Wloc = (__bf16*)take((size_t)A_ * LF_ * 2);
  w.W1p  = (__bf16*)take((size_t)P_ * 96 * 2);
  w.W2p  = (__bf16*)take((size_t)P_ * P_ * 2);
  w.tgtb = (__bf16*)take((size_t)B_ * TDEC * 96 * 2);
  w.x1b  = (__bf16*)take((size_t)B_ * TDEC * P_ * 2);
  w.preb = (__bf16*)take((size_t)TDEC * B_ * P_ * 2);
  w.proc = (float*)take((size_t)B_ * TENC * A_ * 4);
  w.xatt = (__bf16*)take((size_t)B_ * KA_ * 2);
  w.xgen = (__bf16*)take((size_t)B_ * KG_ * 2);
  w.locf = (__bf16*)take((size_t)B_ * TENC * LF_ * 2);
  w.g1   = (float*)take((size_t)B_ * G4_ * 4);
  w.g2   = (float*)take((size_t)B_ * G4_ * 4);
  w.locp = (float*)take((size_t)B_ * TENC * A_ * 4);
  w.hatt = (float*)take((size_t)B_ * D_ * 4);
  w.catt = (float*)take((size_t)B_ * D_ * 4);
  w.hgen = (float*)take((size_t)B_ * D_ * 4);
  w.cgen = (float*)take((size_t)B_ * D_ * 4);
  w.ctx  = (float*)take((size_t)B_ * E_ * 4);
  w.wprev = (float*)take((size_t)B_ * TENC * 4);
  w.wcum  = (float*)take((size_t)B_ * TENC * 4);

  k_init     <<<NWG_, BLK_, 0, stream>>>(w);
  k_convert  <<<1024, BLK_, 0, stream>>>(w, in);
  k_build_tgt<<<1024, BLK_, 0, stream>>>(w, in);
  k_procmem  <<<3200, BLK_, 0, stream>>>(w, in);
  k_prenet1  <<<512,  BLK_, 0, stream>>>(w, in);
  k_prenet2  <<<512,  BLK_, 0, stream>>>(w, in);
  k_decoder  <<<NWG_, BLK_, 0, stream>>>(w, in, out);
}
